// BitLinear_798863917295
// MI455X (gfx1250) — 8-bit matrix operand order repaired and checked by reading, compiled, not run on hardware
//
#include <hip/hip_runtime.h>
#include <cstdint>

typedef __attribute__((ext_vector_type(8))) int v8i;

#define EPSQ 1e-5f

__global__ __launch_bounds__(256)
void quant_weight_kernel(const float* __restrict__ w, int8_t* __restrict__ wq,
                         float* __restrict__ wscale, int I) {
    __shared__ float red[256];
    const int o = blockIdx.x;
    const float* row = w + (size_t)o * I;
    float s = 0.0f;
    for (int i = threadIdx.x; i < I; i += 256) s += fabsf(row[i]);
    red[threadIdx.x] = s;
    __syncthreads();
    for (int d = 128; d > 0; d >>= 1) {
        if (threadIdx.x < d) red[threadIdx.x] += red[threadIdx.x + d];
        __syncthreads();
    }
    float scale = fmaxf(red[0] / (float)I, EPSQ);
    if (threadIdx.x == 0) wscale[o] = scale;
    int8_t* qrow = wq + (size_t)o * I;
    for (int i = threadIdx.x; i < I; i += 256) {
        float q = rintf(row[i] / scale);
        q = fminf(fmaxf(q, -1.0f), 1.0f);
        qrow[i] = (int8_t)q;
    }
}

__global__ __launch_bounds__(256)
void quant_x_kernel(const float* __restrict__ x, int8_t* __restrict__ xq,
                    float* __restrict__ xscale, int I) {
    __shared__ float red[256];
    const int m = blockIdx.x;
    const float* row = x + (size_t)m * I;
    float amax = 0.0f;
    for (int i = threadIdx.x; i < I; i += 256) amax = fmaxf(amax, fabsf(row[i]));
    red[threadIdx.x] = amax;
    __syncthreads();
    for (int d = 128; d > 0; d >>= 1) {
        if (threadIdx.x < d) red[threadIdx.x] = fmaxf(red[threadIdx.x], red[threadIdx.x + d]);
        __syncthreads();
    }
    float scale = fmaxf(red[0] * (1.0f / 127.0f), EPSQ);
    if (threadIdx.x == 0) xscale[m] = scale;
    int8_t* qrow = xq + (size_t)m * I;
    for (int i = threadIdx.x; i < I; i += 256) {
        float q = rintf(row[i] / scale);
        q = fminf(fmaxf(q, -127.0f), 127.0f);
        qrow[i] = (int8_t)q;
    }
}


__device__ __forceinline__ v8i load_a_frag(const int8_t* p) {
    int2 t0 = *(const int2*)(p);
    int2 t1 = *(const int2*)(p + 16);
    int2 t2 = *(const int2*)(p + 32);
    int2 t3 = *(const int2*)(p + 48);
    v8i r;
    r[0] = t0.x; r[1] = t0.y; r[2] = t1.x; r[3] = t1.y;
    r[4] = t2.x; r[5] = t2.y; r[6] = t3.x; r[7] = t3.y;
    return r;
}

__device__ __forceinline__ v8i load_b_frag(const int8_t* p) {
    int2 t0 = *(const int2*)(p);
    int2 t1 = *(const int2*)(p + 16);
    int2 t2 = *(const int2*)(p + 32);
    int2 t3 = *(const int2*)(p + 48);
    v8i r;
    r[0] = t0.x; r[1] = t0.y; r[2] = t1.x; r[3] = t1.y;
    r[4] = t2.x; r[5] = t2.y; r[6] = t3.x; r[7] = t3.y;
    return r;
}

__global__ __launch_bounds__(256)
void bitlinear_gemm_kernel(const int8_t* __restrict__ Aq,
                           const int8_t* __restrict__ Bq,
                           const float* __restrict__ xscale,
                           const float* __restrict__ wscale,
                           const float* __restrict__ bias,
                           float* __restrict__ out,
                           int M, int N, int K) {
    const int lane = threadIdx.x & 31;
    const int wave = threadIdx.x >> 5;
    const int wm   = wave & 1;
    const int wn   = wave >> 1;
    const int l15  = lane & 15;
    const int hi   = lane >> 4;

    const int mBase = blockIdx.y * 64  + wm * 32;
    const int nBase = blockIdx.x * 256 + wn * 64;

    const size_t aRow0 = (size_t)(mBase + l15) * K + hi * 8;
    const size_t aRow1 = (size_t)(mBase + 16 + l15) * K + hi * 8;
    size_t bRow[4];
#pragma unroll
    for (int nt = 0; nt < 4; ++nt)
        bRow[nt] = (size_t)(nBase + nt * 16 + l15) * K + hi * 8;

    v8i acc[2][4];
#pragma unroll
    for (int mt = 0; mt < 2; ++mt)
#pragma unroll
        for (int nt = 0; nt < 4; ++nt)
            acc[mt][nt] = (v8i)0;

    for (int k0 = 0; k0 < K; k0 += 64) {
        const v8i a0 = load_a_frag(Aq + aRow0 + k0);
        const v8i a1 = load_a_frag(Aq + aRow1 + k0);
        v8i b[4];
#pragma unroll
        for (int nt = 0; nt < 4; ++nt)
            b[nt] = load_b_frag(Bq + bRow[nt] + k0);

#pragma unroll
        for (int nt = 0; nt < 4; ++nt) {
            acc[0][nt] = __builtin_amdgcn_wmma_i32_16x16x64_iu8(
                true, a0, true, b[nt], acc[0][nt], false, false);
            acc[1][nt] = __builtin_amdgcn_wmma_i32_16x16x64_iu8(
                true, a1, true, b[nt], acc[1][nt], false, false);
        }
    }

#pragma unroll
    for (int mt = 0; mt < 2; ++mt) {
        const int mrow0 = mBase + mt * 16 + hi * 8;
#pragma unroll
        for (int nt = 0; nt < 4; ++nt) {
            const int n = nBase + nt * 16 + l15;
            const float ws = wscale[n];
            const float bz = bias[n];
#pragma unroll
            for (int j = 0; j < 8; ++j) {
                const int m = mrow0 + j;
                const float v = (float)acc[mt][nt][j] * (xscale[m] * ws) + bz;
                out[(size_t)m * N + n] = v;
            }
        }
    }
}

extern "C" void kernel_launch(void* const* d_in, const int* in_sizes, int n_in,
                              void* d_out, int out_size, void* d_ws, size_t ws_size,
                              hipStream_t stream) {
    const float* x    = (const float*)d_in[0];
    const float* w    = (const float*)d_in[1];
    const float* bias = (const float*)d_in[2];
    float* out = (float*)d_out;

    const int O = in_sizes[2];
    const int I = in_sizes[1] / O;
    const int M = in_sizes[0] / I;

    int8_t* wq = (int8_t*)d_ws;
    int8_t* xq = wq + (size_t)O * I;
    float* wscale = (float*)(xq + (size_t)M * I);
    float* xscale = wscale + O;

    quant_weight_kernel<<<O, 256, 0, stream>>>(w, wq, wscale, I);
    quant_x_kernel<<<M, 256, 0, stream>>>(x, xq, xscale, I);

    dim3 grid(O / 256, M / 64);
    bitlinear_gemm_kernel<<<grid, 256, 0, stream>>>(xq, wq, xscale, wscale, bias,
                                                    out, M, O, I);
}
